// Discriminator_89232240542134
// MI455X (gfx1250) — matrix operand order repaired and checked by reading, not compiled, not run on hardware
//
#include <hip/hip_runtime.h>
#include <hip/hip_bf16.h>
#include <math.h>

#define LEAK 0.2f
#define BN_EPS 1e-4f

typedef float v8f __attribute__((ext_vector_type(8)));
typedef __bf16 v16bf __attribute__((ext_vector_type(16)));
typedef unsigned short u16v8 __attribute__((ext_vector_type(8)));
typedef unsigned short u16v4 __attribute__((ext_vector_type(4)));
typedef unsigned int u32x4 __attribute__((ext_vector_type(4)));
typedef int i32x8 __attribute__((ext_vector_type(8)));
typedef int i32x4 __attribute__((ext_vector_type(4)));

union Frag16 { u16v8 h[2]; v16bf v; };

__device__ inline unsigned short f2bf(float f) {
  unsigned int u = __float_as_uint(f);
  u += 0x7FFFu + ((u >> 16) & 1u);   // round-to-nearest-even
  return (unsigned short)(u >> 16);
}
__device__ inline float bf2f(unsigned short h) {
  return __uint_as_float(((unsigned int)h) << 16);
}

// --------------------------------------------------------------- TDM issue
// Builds a D# per CDNA5 ISA §8 and issues TENSOR_LOAD_TO_LDS.
// Dims: x = tile0 (contig elems), y = tile1 (stride s0), z = tile2 (stride s1),
// zz = tile3 (stride s2). All dims/strides in 2-byte element units.
__device__ inline void tdm_load(unsigned lds, unsigned long long ga,
                                unsigned tile0, unsigned tile1, unsigned tile2,
                                unsigned tile3, unsigned long long s0,
                                unsigned long long s1, unsigned long long s2) {
  u32x4 g0;
  g0.x = 1u;                                   // count=1, user mode
  g0.y = lds;                                  // lds_addr (bytes)
  g0.z = (unsigned)ga;                         // global_addr[31:0]
  g0.w = (unsigned)((ga >> 32) & 0x01FFFFFFull) | 0x80000000u;  // ga[56:32] | type=2

  unsigned td0 = tile0, td1 = tile1;           // tensor dims == tile dims (no OOB)
  i32x8 g1;
  g1[0] = (int)(1u << 16);                     // data_size = 2 bytes
  g1[1] = (int)((td0 & 0xFFFFu) << 16);
  g1[2] = (int)((td0 >> 16) | ((td1 & 0xFFFFu) << 16));
  g1[3] = (int)((td1 >> 16) | (tile0 << 16));  // tile_dim0
  g1[4] = (int)(tile1 | (tile2 << 16));        // tile_dim1, tile_dim2
  g1[5] = (int)(unsigned)s0;                   // tensor_dim0_stride[31:0]
  g1[6] = (int)(((unsigned)(s0 >> 32) & 0xFFFFu) | (((unsigned)s1 & 0xFFFFu) << 16));
  g1[7] = (int)(unsigned)(s1 >> 16);

  i32x4 g2;
  g2[0] = (int)tile2;                          // tensor_dim2
  g2[1] = (int)tile3;                          // tensor_dim3
  g2[2] = (int)(unsigned)s2;                   // tensor_dim2_stride[31:0]
  g2[3] = (int)(((unsigned)(s2 >> 32) & 0xFFFFu) | (tile3 << 16));  // tile_dim3

  i32x4 g3;
  g3[0] = 0; g3[1] = 0; g3[2] = 0; g3[3] = 0;

  i32x8 gx;                                    // extension words (zero-filled)
  gx[0] = 0; gx[1] = 0; gx[2] = 0; gx[3] = 0;
  gx[4] = 0; gx[5] = 0; gx[6] = 0; gx[7] = 0;

  __builtin_amdgcn_tensor_load_to_lds(g0, g1, g2, g3, gx, 0);
}

// ---------------------------------------------------------------- zero
__global__ void zero_f4(float4* __restrict__ p, long n4) {
  long i = (long)blockIdx.x * blockDim.x + threadIdx.x;
  if (i < n4) p[i] = make_float4(0.f, 0.f, 0.f, 0.f);
}

// ---------------------------------------------------------------- scatter (densify accumulate)
__global__ void scatter_k(const float* __restrict__ feat, const int* __restrict__ coords,
                          float* __restrict__ g0, float* __restrict__ cnt, int P) {
  int idx = blockIdx.x * blockDim.x + threadIdx.x;
  if (idx >= P * 32) return;
  int p = idx >> 5, c = idx & 31;
  int b = coords[p * 4 + 0], x = coords[p * 4 + 1];
  int y = coords[p * 4 + 2], z = coords[p * 4 + 3];
  long lin = (((long)b * 144 + x) * 64 + y) * 160 + z;
  atomicAdd(&g0[lin * 32 + c], feat[idx]);
  if (c == 0) atomicAdd(&cnt[lin], 1.0f);
}

// ------------------------------------------- densify finalize -> bf16 grid
__global__ void densify_k(const float* __restrict__ g0sum, const float* __restrict__ cnt,
                          unsigned short* __restrict__ g0bf, long nvox) {
  long idx = (long)blockIdx.x * blockDim.x + threadIdx.x;
  if (idx >= nvox * 8) return;
  long v = idx >> 3; int j = (int)(idx & 7);
  float inv = 1.0f / fmaxf(cnt[v], 1.0f);
  const float4 t = *(const float4*)(g0sum + v * 32 + j * 4);
  u16v4 h;
  h.x = f2bf(t.x * inv); h.y = f2bf(t.y * inv);
  h.z = f2bf(t.z * inv); h.w = f2bf(t.w * inv);
  *(u16v4*)(g0bf + v * 32 + j * 4) = h;
}

// ------------------------------------------- weight prep: f32 [K][N] -> bf16 [N][K]
template <int COUT>
__global__ void wprep_k(const float* __restrict__ w, int ktot, unsigned short* __restrict__ wt) {
  int idx = blockIdx.x * blockDim.x + threadIdx.x;
  if (idx >= ktot * COUT) return;
  int k = idx / COUT, n = idx % COUT;
  wt[n * ktot + k] = f2bf(w[idx]);
}

// ---------------------------------------------------------------- stride-2 2x2x2 conv as WMMA GEMM
// WG = 128 threads (4 waves) computes a (TW x TH x TD)=64-voxel output sub-tile x all COUT.
// A/B panels are DMA'd into double-buffered LDS by the Tensor Data Mover (wave 0 issues),
// overlapped with WMMA on the previous K-chunk via s_wait_tensorcnt.
template <int CIN, int COUT, int DI, int HI, int WI, int DOo, int HOo, int WOo,
          int TW, int TH, int TD>
__global__ __launch_bounds__(128) void conv_k(const unsigned short* __restrict__ xin,
                                              const float* __restrict__ maskin,
                                              const unsigned short* __restrict__ wt,
                                              float* __restrict__ xout,
                                              float* __restrict__ maskout) {
  constexpr int NT = COUT / 16;
  constexpr int KC = (8 * CIN) / 32;
  constexpr int KTOT = 8 * CIN;
  __shared__ unsigned short sA[2][64 * 32];     // [buf][row64][k32]
  __shared__ unsigned short sB[2][COUT * 32];   // [buf][n][k32]

  const int tid = threadIdx.x;
  const int wave = tid >> 5;
  const int lane = tid & 31;
  const int hi = lane >> 4;
  const int col = lane & 15;

  int t0 = blockIdx.x;
  const int wb = t0 % (WOo / TW); t0 /= (WOo / TW);
  const int hb = t0 % (HOo / TH); t0 /= (HOo / TH);
  const int db = t0 % (DOo / TD); const int bq = t0 / (DOo / TD);

  // downsampled mask for this tile's 64 output voxels
  if (tid < 64) {
    const int dI = tid / (TH * TW), hI = (tid / TW) % TH, wI = tid % TW;
    const int dq = db * TD + dI, hq = hb * TH + hI, wq = wb * TW + wI;
    float m = 0.0f;
#pragma unroll
    for (int tap = 0; tap < 8; ++tap) {
      int kd = tap >> 2, kh = (tap >> 1) & 1, kw = tap & 1;
      long vin = (((long)bq * DI + (2 * dq + kd)) * HI + (2 * hq + kh)) * WI + (2 * wq + kw);
      m = fmaxf(m, maskin[vin]);     // level-0 passes counts; >0 handles both
    }
    long o = (((long)bq * DOo + dq) * HOo + hq) * WOo + wq;
    maskout[o] = (m > 0.0f) ? 1.0f : 0.0f;
  }

  auto issueA = [&](int kc, int bsel) {
    const int t = (kc * 32) / CIN, cb = (kc * 32) % CIN;
    const int kd = t >> 2, kh = (t >> 1) & 1, kw = t & 1;
    const long vin0 = (((long)bq * DI + (2 * (db * TD) + kd)) * HI + (2 * (hb * TH) + kh)) * WI
                      + (2 * (wb * TW) + kw);
    unsigned long long ga = (unsigned long long)(size_t)xin
                            + 2ull * ((unsigned long long)vin0 * CIN + (unsigned)cb);
    tdm_load((unsigned)(size_t)&sA[bsel][0], ga,
             32u, (unsigned)TW, (unsigned)TH, (unsigned)(TD > 1 ? TD : 0),
             (unsigned long long)(2 * CIN), (unsigned long long)(2 * WI * CIN),
             (unsigned long long)(2ll * HI * WI * CIN));
  };
  auto issueB = [&](int kc, int bsel) {
    unsigned long long ga = (unsigned long long)(size_t)wt + 2ull * (unsigned)(kc * 32);
    tdm_load((unsigned)(size_t)&sB[bsel][0], ga,
             32u, (unsigned)COUT, 0u, 0u,
             (unsigned long long)KTOT, 0ull, 0ull);
  };

  if (tid < 32) { issueA(0, 0); issueB(0, 0); }   // prologue: chunk 0

  v8f acc[NT];
#pragma unroll
  for (int nt = 0; nt < NT; ++nt) acc[nt] = (v8f)0.0f;

  const int arow = wave * 16 + col;   // A row for this lane

  for (int kc = 0; kc < KC; ++kc) {
    if (kc + 1 < KC) {
      if (tid < 32) { issueA(kc + 1, (kc + 1) & 1); issueB(kc + 1, (kc + 1) & 1); }
      __builtin_amdgcn_s_wait_tensorcnt(2);   // chunk kc complete (in-order)
    } else {
      __builtin_amdgcn_s_wait_tensorcnt(0);
    }
    __syncthreads();

    const unsigned short* A = sA[kc & 1];
    const unsigned short* B = sB[kc & 1];
    Frag16 a;
    a.h[0] = *(const u16v8*)(&A[arow * 32 + hi * 8]);
    a.h[1] = *(const u16v8*)(&A[arow * 32 + hi * 8 + 16]);
#pragma unroll
    for (int nt = 0; nt < NT; ++nt) {
      Frag16 b;
      b.h[0] = *(const u16v8*)(&B[(nt * 16 + col) * 32 + hi * 8]);
      b.h[1] = *(const u16v8*)(&B[(nt * 16 + col) * 32 + hi * 8 + 16]);
      acc[nt] = __builtin_amdgcn_wmma_f32_16x16x32_bf16(
          false, a.v, false, b.v, (short)0, acc[nt], false, false);
    }
    __syncthreads();
  }

  // C/D layout: VGPR r -> M = r + hi*8 within the wave's 16-row tile; N = col
#pragma unroll
  for (int nt = 0; nt < NT; ++nt) {
#pragma unroll
    for (int r = 0; r < 8; ++r) {
      const int m = wave * 16 + r + hi * 8;               // row in 64-voxel tile
      const int dI = m / (TH * TW), hI = (m / TW) % TH, wI = m % TW;
      const long o = (((long)bq * DOo + (db * TD + dI)) * HOo + (hb * TH + hI)) * WOo
                     + (wb * TW + wI);
      xout[o * COUT + nt * 16 + col] = acc[nt][r];
    }
  }
}

// ---------------------------------------------------------------- BN stats: sum, sumsq, n_active
template <int C>
__global__ void stats_k(const float* __restrict__ x, const float* __restrict__ mask,
                        long nvox, float* __restrict__ stats) {
  const int tid = threadIdx.x;     // blockDim == 2*C
  const int c = tid % C;
  const int r = tid / C;
  float s = 0.f, ss = 0.f, mn = 0.f;
  for (long v = (long)blockIdx.x * 2 + r; v < nvox; v += (long)gridDim.x * 2) {
    float val = x[v * C + c];
    s += val; ss += val * val;
    if (c == 0) mn += mask[v];
  }
  __shared__ float sS[C];
  __shared__ float sQ[C];
  if (r == 0) { sS[c] = s; sQ[c] = ss; }
  __syncthreads();
  if (r == 1) { sS[c] += s; sQ[c] += ss; }
  __syncthreads();
  if (tid < C) {
    atomicAdd(&stats[c], sS[c]);
    atomicAdd(&stats[C + c], sQ[c]);
  }
  if (c == 0) atomicAdd(&stats[2 * C], mn);
}

// ------------------------------- BN + LeakyReLU + mask -> bf16 activations
template <int C>
__global__ void bnapply_k(const float* __restrict__ xraw, const float* __restrict__ mask,
                          const float* __restrict__ stats,
                          const float* __restrict__ gamma, const float* __restrict__ beta,
                          unsigned short* __restrict__ xbf, long nvox) {
  long idx = (long)blockIdx.x * blockDim.x + threadIdx.x;
  if (idx >= nvox * C) return;
  long v = idx / C; int c = (int)(idx % C);
  float n = fmaxf(stats[2 * C], 1.0f);
  float mean = stats[c] / n;
  float var = stats[C + c] / n - mean * mean;
  float A = gamma[c] * rsqrtf(var + BN_EPS);
  float Bs = beta[c] - mean * A;
  float y = xraw[idx] * A + Bs;
  y = (y > 0.f) ? y : LEAK * y;
  xbf[idx] = f2bf(y * mask[v]);
}

// ---------------------------------------------------------------- final conv (Cout=1) + sigmoid*mask
__global__ __launch_bounds__(128) void conv4_k(const unsigned short* __restrict__ x3,
                                               const float* __restrict__ mask3,
                                               const float* __restrict__ w4,
                                               float* __restrict__ out) {
  int o = blockIdx.x * 4 + (threadIdx.x >> 5);   // one wave per output voxel; 720 total
  int lane = threadIdx.x & 31;
  int wq = o % 10, hq = (o / 10) % 4, dq = (o / 40) % 9, bq = o / 360;
  int vin8[8];
#pragma unroll
  for (int tap = 0; tap < 8; ++tap) {
    int kd = tap >> 2, kh = (tap >> 1) & 1, kw = tap & 1;
    vin8[tap] = (((bq * 18) + (2 * dq + kd)) * 8 + (2 * hq + kh)) * 20 + (2 * wq + kw);
  }
  float s = 0.f;
  for (int k = lane; k < 1024; k += 32) {       // K = 8 taps * 128 ch
    int tap = k >> 7, c = k & 127;
    s += bf2f(x3[(long)vin8[tap] * 128 + c]) * w4[k];
  }
#pragma unroll
  for (int off = 16; off > 0; off >>= 1) s += __shfl_down(s, off, 32);
  if (lane == 0) {
    float m = 0.f;
#pragma unroll
    for (int tap = 0; tap < 8; ++tap) m = fmaxf(m, mask3[vin8[tap]]);
    float sig = 1.0f / (1.0f + __expf(-s));
    out[o] = (m > 0.f) ? sig : 0.f;
  }
}

// ---------------------------------------------------------------- launcher
extern "C" void kernel_launch(void* const* d_in, const int* in_sizes, int n_in,
                              void* d_out, int out_size, void* d_ws, size_t ws_size,
                              hipStream_t stream) {
  (void)in_sizes; (void)n_in; (void)out_size; (void)ws_size;
  const float* feat = (const float*)d_in[0];
  const int* coords = (const int*)d_in[1];
  const float* w1 = (const float*)d_in[2];
  const float* g1 = (const float*)d_in[3];
  const float* b1 = (const float*)d_in[4];
  const float* w2 = (const float*)d_in[5];
  const float* g2 = (const float*)d_in[6];
  const float* b2 = (const float*)d_in[7];
  const float* w3 = (const float*)d_in[8];
  const float* g3 = (const float*)d_in[9];
  const float* b3 = (const float*)d_in[10];
  const float* w4 = (const float*)d_in[11];

  const long NV0 = 2L * 144 * 64 * 160;   // 2,949,120
  const long NV1 = 2L * 72 * 32 * 80;     //   368,640
  const long NV2 = 2L * 36 * 16 * 40;     //    46,080
  const long NV3 = 2L * 18 * 8 * 20;      //     5,760
  const int  P   = 300000;

  char* ws = (char*)d_ws;
  size_t off = 0;
  auto alloc = [&](size_t bytes) -> void* {
    void* p = ws + off;
    off = (off + bytes + 255) & ~(size_t)255;
    return p;
  };
  float*          g0sum = (float*)alloc((size_t)NV0 * 32 * 4);
  float*          cnt0  = (float*)alloc((size_t)NV0 * 4);
  unsigned short* g0bf  = (unsigned short*)alloc((size_t)NV0 * 32 * 2);
  unsigned short* wt1   = (unsigned short*)alloc((size_t)256 * 64 * 2);
  unsigned short* wt2   = (unsigned short*)alloc((size_t)512 * 96 * 2);
  unsigned short* wt3   = (unsigned short*)alloc((size_t)768 * 128 * 2);
  float*          xr1   = (float*)alloc((size_t)NV1 * 64 * 4);
  unsigned short* xb1   = (unsigned short*)alloc((size_t)NV1 * 64 * 2);
  float*          mask1 = (float*)alloc((size_t)NV1 * 4);
  float*          xr2   = (float*)alloc((size_t)NV2 * 96 * 4);
  unsigned short* xb2   = (unsigned short*)alloc((size_t)NV2 * 96 * 2);
  float*          mask2 = (float*)alloc((size_t)NV2 * 4);
  float*          xr3   = (float*)alloc((size_t)NV3 * 128 * 4);
  unsigned short* xb3   = (unsigned short*)alloc((size_t)NV3 * 128 * 2);
  float*          mask3 = (float*)alloc((size_t)NV3 * 4);
  float*          st1   = (float*)alloc(512 * 4);
  float*          st2   = (float*)alloc(512 * 4);
  float*          st3   = (float*)alloc(512 * 4);

  // zero scatter accumulators + stats (st1..st3 contiguous)
  {
    long n4 = NV0 * 32 / 4;
    zero_f4<<<(int)((n4 + 255) / 256), 256, 0, stream>>>((float4*)g0sum, n4);
    n4 = NV0 / 4;
    zero_f4<<<(int)((n4 + 255) / 256), 256, 0, stream>>>((float4*)cnt0, n4);
    zero_f4<<<2, 256, 0, stream>>>((float4*)st1, (3 * 512) / 4);
  }

  // weight transpose + bf16
  wprep_k<64><<<(256 * 64) / 256, 256, 0, stream>>>(w1, 256, wt1);
  wprep_k<96><<<(512 * 96) / 256, 256, 0, stream>>>(w2, 512, wt2);
  wprep_k<128><<<(768 * 128) / 256, 256, 0, stream>>>(w3, 768, wt3);

  scatter_k<<<(P * 32 + 255) / 256, 256, 0, stream>>>(feat, coords, g0sum, cnt0, P);
  densify_k<<<(int)((NV0 * 8 + 255) / 256), 256, 0, stream>>>(g0sum, cnt0, g0bf, NV0);

  conv_k<32, 64, 144, 64, 160, 72, 32, 80, 16, 4, 1>
      <<<(int)(NV1 / 64), 128, 0, stream>>>(g0bf, cnt0, wt1, xr1, mask1);
  stats_k<64><<<256, 128, 0, stream>>>(xr1, mask1, NV1, st1);
  bnapply_k<64><<<(int)((NV1 * 64 + 255) / 256), 256, 0, stream>>>(xr1, mask1, st1, g1, b1, xb1, NV1);

  conv_k<64, 96, 72, 32, 80, 36, 16, 40, 8, 8, 1>
      <<<(int)(NV2 / 64), 128, 0, stream>>>(xb1, mask1, wt2, xr2, mask2);
  stats_k<96><<<256, 192, 0, stream>>>(xr2, mask2, NV2, st2);
  bnapply_k<96><<<(int)((NV2 * 96 + 255) / 256), 256, 0, stream>>>(xr2, mask2, st2, g2, b2, xb2, NV2);

  conv_k<96, 128, 36, 16, 40, 18, 8, 20, 4, 8, 2>
      <<<(int)(NV3 / 64), 128, 0, stream>>>(xb2, mask2, wt3, xr3, mask3);
  stats_k<128><<<256, 256, 0, stream>>>(xr3, mask3, NV3, st3);
  bnapply_k<128><<<(int)((NV3 * 128 + 255) / 256), 256, 0, stream>>>(xr3, mask3, st3, g3, b3, xb3, NV3);

  conv4_k<<<720 / 4, 128, 0, stream>>>(xb3, mask3, w4, (float*)d_out);
}
